// GlobalSelfAttentionLayer_34299608826530
// MI455X (gfx1250) — compile-verified, matrix operand order checked by reading, not run on hardware
//
#include <hip/hip_runtime.h>
#include <hip/hip_bf16.h>
#include <math.h>

typedef __attribute__((ext_vector_type(16))) __bf16 v16bf;
typedef __attribute__((ext_vector_type(8)))  __bf16 v8bf;
typedef __attribute__((ext_vector_type(8)))  float  v8f;
typedef __attribute__((ext_vector_type(4)))  int    v4i;

#define NROWS 8192
#define MROWS 8192
#define FDIM  512
#define HDIM  512
#define VIEWS 3
#define BN 64
#define BM 64
#define EPSN 1e-12f

// ---- CDNA5 async global->LDS path (ASYNCcnt), with safe fallback ----
#if defined(__HIP_DEVICE_COMPILE__) && \
    __has_builtin(__builtin_amdgcn_global_load_async_to_lds_b128)
#define HAVE_ASYNC 1
#else
#define HAVE_ASYNC 0
#endif

#if HAVE_ASYNC
typedef __attribute__((address_space(1))) v4i gv4i;   // global int4
typedef __attribute__((address_space(3))) v4i lv4i;   // LDS int4
__device__ __forceinline__ void async_b128(const void* g, void* l) {
  __builtin_amdgcn_global_load_async_to_lds_b128((gv4i*)(void*)g, (lv4i*)l, 0, 0);
}
#if __has_builtin(__builtin_amdgcn_s_wait_asynccnt)
#define WAIT_ASYNC(n) __builtin_amdgcn_s_wait_asynccnt(n)
#else
#define WAIT_ASYNC(n) asm volatile("s_wait_asynccnt %0" ::"n"(n) : "memory")
#endif
#else
#define WAIT_ASYNC(n) ((void)0)
#endif

__device__ __forceinline__ v16bf ld_frag(const __bf16* p) {
  v8bf lo = *(const v8bf*)(p);
  v8bf hi = *(const v8bf*)(p + 8);
  v16bf r;
#pragma unroll
  for (int i = 0; i < 8; ++i) { r[i] = lo[i]; r[i + 8] = hi[i]; }
  return r;
}

__device__ __forceinline__ v8f wmma_bf16(v16bf a, v16bf b, v8f c) {
  return __builtin_amdgcn_wmma_f32_16x16x32_bf16(false, a, false, b, (short)0, c,
                                                 false, false);
}

// C[r,h] = elu(sum_f X[r,f] * W[h,f]) stored as bf16.  64x64 output tile per block.
__global__ __launch_bounds__(256)
void elu_gemm_bf16(const float* __restrict__ X, const float* __restrict__ W,
                   __bf16* __restrict__ out) {
  __shared__ __bf16 Xt[64 * 32];
  __shared__ __bf16 Wt[64 * 32];
  const int tid = threadIdx.x;
  const int lane = tid & 31, w = tid >> 5;
  const int rb = (w >> 1) * 16;
  const int cb = (w & 1) * 32;
  const int l15 = lane & 15;
  const int kh = (lane >> 4) * 16;
  const int r0 = blockIdx.x * 64;
  const int h0 = blockIdx.y * 64;
  v8f acc0 = {}, acc1 = {};
  for (int kc = 0; kc < FDIM; kc += 32) {
    __syncthreads();
#pragma unroll
    for (int i = 0; i < 8; ++i) {
      int e = tid + 256 * i;
      int rr = e >> 5, cc = e & 31;
      Xt[rr * 32 + cc] = (__bf16)X[(size_t)(r0 + rr) * FDIM + kc + cc];
      Wt[rr * 32 + cc] = (__bf16)W[(size_t)(h0 + rr) * FDIM + kc + cc];
    }
    __syncthreads();
    v16bf a  = ld_frag(&Xt[(rb + l15) * 32 + kh]);
    v16bf b0 = ld_frag(&Wt[(cb + l15) * 32 + kh]);
    v16bf b1 = ld_frag(&Wt[(cb + 16 + l15) * 32 + kh]);
    acc0 = wmma_bf16(a, b0, acc0);
    acc1 = wmma_bf16(a, b1, acc1);
  }
  const int hi = (lane >> 4) * 8;
#pragma unroll
  for (int i = 0; i < 8; ++i) {
    int row = r0 + rb + i + hi;
    float x0 = acc0[i]; x0 = x0 > 0.f ? x0 : (expf(x0) - 1.f);
    float x1 = acc1[i]; x1 = x1 > 0.f ? x1 : (expf(x1) - 1.f);
    out[(size_t)row * HDIM + h0 + cb + l15]      = (__bf16)x0;
    out[(size_t)row * HDIM + h0 + cb + 16 + l15] = (__bf16)x1;
  }
}

// zs [V][M][F] fp32 -> zsbT [V][F][M] bf16 (64x64 tiles via LDS)
__global__ __launch_bounds__(256)
void transpose_bf16(const float* __restrict__ zs, __bf16* __restrict__ zsbT) {
  __shared__ __bf16 tile[64][65];
  const int m0 = blockIdx.x * 64, f0 = blockIdx.y * 64, v = blockIdx.z;
  const float* src = zs + (size_t)v * MROWS * FDIM;
  __bf16* dst = zsbT + (size_t)v * FDIM * MROWS;
#pragma unroll
  for (int i = 0; i < 16; ++i) {
    int e = threadIdx.x + 256 * i;
    int r = e >> 6, c = e & 63;
    tile[r][c] = (__bf16)src[(size_t)(m0 + r) * FDIM + f0 + c];
  }
  __syncthreads();
#pragma unroll
  for (int i = 0; i < 16; ++i) {
    int e = threadIdx.x + 256 * i;
    int fr = e >> 6, mc = e & 63;
    dst[(size_t)(f0 + fr) * MROWS + m0 + mc] = tile[mc][fr];
  }
}

// Stage one (k-tile, zvT-tile) pair into LDS (async on CDNA5, sync fallback).
__device__ __forceinline__ void stage_tile(const __bf16* __restrict__ kv,
                                           const __bf16* __restrict__ zvT,
                                           int m0, __bf16* kt, __bf16* zvt,
                                           int tid) {
#pragma unroll
  for (int i = 0; i < 16; ++i) {
    int c = tid + 256 * i;              // 4096 chunks of 8 bf16 (b128)
    int r = c >> 6, c8 = (c & 63) * 8;
#if HAVE_ASYNC
    async_b128(&kv[(size_t)(m0 + r) * HDIM + c8], &kt[r * HDIM + c8]);
#else
    *(v8bf*)&kt[r * HDIM + c8] = *(const v8bf*)&kv[(size_t)(m0 + r) * HDIM + c8];
#endif
  }
#pragma unroll
  for (int i = 0; i < 16; ++i) {
    int c = tid + 256 * i;
    int f = c >> 3, c8 = (c & 7) * 8;
#if HAVE_ASYNC
    async_b128(&zvT[(size_t)f * MROWS + m0 + c8], &zvt[f * BM + c8]);
#else
    *(v8bf*)&zvt[f * BM + c8] = *(const v8bf*)&zvT[(size_t)f * MROWS + m0 + c8];
#endif
  }
}

// Fused flash-style pass: S = q@k^T (never materialized in HBM), rowsum(S^2)
// and acc += S@zv in one sweep over M; out += acc / max(||row||, eps).
// Double-buffered LDS tiles with CDNA5 async global->LDS loads.
__global__ __launch_bounds__(256)
void fused_attn(const float* __restrict__ common_z,
                const __bf16* __restrict__ qb,
                const __bf16* __restrict__ kb,
                const __bf16* __restrict__ zsbT,
                float* __restrict__ out) {
  extern __shared__ char smem[];
  const size_t TILE = (size_t)BM * HDIM;   // 32768 bf16 = 64 KB
  __bf16* ktb[2];
  __bf16* zvb[2];
  ktb[0] = (__bf16*)smem;
  ktb[1] = ktb[0] + TILE;
  zvb[0] = ktb[1] + TILE;
  zvb[1] = zvb[0] + TILE;
  __bf16* sS = zvb[1] + TILE;              // [BN][BM]
  float* sumsq = (float*)(sS + BN * BM);   // [BN]

  const int tid = threadIdx.x;
  const int lane = tid & 31, w = tid >> 5;
  const int rg = w >> 1;                   // 0..3 : 16-row group
  const int ch = w & 1;                    // column half
  const int rb = rg * 16;
  const int hi = (lane >> 4) * 8;
  const int l15 = lane & 15;
  const int kh = (lane >> 4) * 16;
  const int n0 = blockIdx.x * BN;
  const int NIT = MROWS / BM;              // 128

  // q fragments for this wave's 16-row strip over full K=512 (in VGPRs)
  v16bf qf[16];
#pragma unroll
  for (int kk = 0; kk < 16; ++kk)
    qf[kk] = ld_frag(&qb[(size_t)(n0 + rb + l15) * HDIM + kk * 32 + kh]);

  for (int v = 0; v < VIEWS; ++v) {
    const __bf16* kv  = kb   + (size_t)v * MROWS * HDIM;
    const __bf16* zvT = zsbT + (size_t)v * FDIM * MROWS;
    if (tid < BN) sumsq[tid] = 0.f;
    v8f acc[16] = {};

    stage_tile(kv, zvT, 0, ktb[0], zvb[0], tid);   // prologue -> buf 0

    for (int it = 0; it < NIT; ++it) {
      const int cur = it & 1;
      const __bf16* kt  = ktb[cur];
      const __bf16* zvt = zvb[cur];
      if (it + 1 < NIT) {
        stage_tile(kv, zvT, (it + 1) * BM, ktb[1 - cur], zvb[1 - cur], tid);
        WAIT_ASYNC(32);   // current tile retired; next tile's 32 loads in flight
      } else {
        WAIT_ASYNC(0);
      }
      __syncthreads();

      // GEMM1: two 16x16 S tiles per wave, K = 512
#pragma unroll
      for (int j = 0; j < 2; ++j) {
        const int mc = (2 * ch + j) * 16;
        v8f s = {};
#pragma unroll
        for (int kk = 0; kk < 16; ++kk) {
          v16bf b = ld_frag(&kt[(mc + l15) * HDIM + kk * 32 + kh]);
          s = wmma_bf16(qf[kk], b, s);
        }
#pragma unroll
        for (int i = 0; i < 8; ++i) {
          float sq = s[i] * s[i];
          sq += __shfl_xor(sq, 1);
          sq += __shfl_xor(sq, 2);
          sq += __shfl_xor(sq, 4);
          sq += __shfl_xor(sq, 8);
          if (l15 == 0) atomicAdd(&sumsq[rb + i + hi], sq);
          sS[(rb + i + hi) * BM + mc + l15] = (__bf16)s[i];
        }
      }
      __syncthreads();

      // GEMM2: acc[16x256 per wave] += S(16x64) @ zv(64x256)
      v16bf a0 = ld_frag(&sS[(rb + l15) * BM + kh]);
      v16bf a1 = ld_frag(&sS[(rb + l15) * BM + 32 + kh]);
#pragma unroll
      for (int t = 0; t < 16; ++t) {
        const int fc = ch * 256 + t * 16;
        v16bf b0 = ld_frag(&zvt[(fc + l15) * BM + kh]);
        v16bf b1 = ld_frag(&zvt[(fc + l15) * BM + 32 + kh]);
        acc[t] = wmma_bf16(a0, b0, acc[t]);
        acc[t] = wmma_bf16(a1, b1, acc[t]);
      }
      __syncthreads();   // all reads of buf[cur] done; safe to overwrite next iter
    }

    // epilogue: scale by 1/max(||row||,eps), accumulate into out
    float inv[8];
#pragma unroll
    for (int i = 0; i < 8; ++i)
      inv[i] = 1.f / fmaxf(sqrtf(sumsq[rb + i + hi]), EPSN);
#pragma unroll
    for (int t = 0; t < 16; ++t) {
      const int fc = ch * 256 + t * 16 + l15;
#pragma unroll
      for (int i = 0; i < 8; ++i) {
        size_t idx = (size_t)(n0 + rb + i + hi) * FDIM + fc;
        float base = (v == 0) ? common_z[idx] : out[idx];
        out[idx] = base + acc[t][i] * inv[i];
      }
    }
    __syncthreads();
  }
}

extern "C" void kernel_launch(void* const* d_in, const int* in_sizes, int n_in,
                              void* d_out, int out_size, void* d_ws, size_t ws_size,
                              hipStream_t stream) {
  (void)in_sizes; (void)n_in; (void)out_size; (void)ws_size;
  const float* common_z = (const float*)d_in[0];   // [N, F]
  const float* zs       = (const float*)d_in[1];   // [V, M, F]
  const float* Wq       = (const float*)d_in[2];   // [H, F]
  const float* Wk       = (const float*)d_in[3];   // [H, F]
  float* out = (float*)d_out;

  char* ws = (char*)d_ws;
  __bf16* qb   = (__bf16*)ws;                                             // 8 MB
  __bf16* kb   = (__bf16*)(ws + (size_t)NROWS * HDIM * 2);                // 24 MB
  __bf16* zsbT = (__bf16*)(ws + (size_t)NROWS * HDIM * 2
                              + (size_t)VIEWS * MROWS * HDIM * 2);        // 24 MB

  dim3 blk(256);
  elu_gemm_bf16<<<dim3(NROWS / 64, HDIM / 64), blk, 0, stream>>>(common_z, Wq, qb);
  for (int v = 0; v < VIEWS; ++v)
    elu_gemm_bf16<<<dim3(MROWS / 64, HDIM / 64), blk, 0, stream>>>(
        zs + (size_t)v * MROWS * FDIM, Wk, kb + (size_t)v * MROWS * HDIM);
  transpose_bf16<<<dim3(MROWS / 64, FDIM / 64, VIEWS), blk, 0, stream>>>(zs, zsbT);

  // double-buffered tiles: 4 x 64KB + S (8KB) + sumsq
  size_t smem = (size_t)4 * BM * HDIM * 2 + (size_t)BN * BM * 2 + BN * 4;
  fused_attn<<<dim3(NROWS / BN), blk, smem, stream>>>(common_z, qb, kb, zsbT, out);
}
